// AttentionBlock_52097953300707
// MI455X (gfx1250) — matrix operand order repaired and checked by reading, not compiled, not run on hardware
//
#include <hip/hip_runtime.h>

// ---------------------------------------------------------------------------
// Attention block for MI455X (gfx1250): bf16 WMMA GEMMs + fp32 softmax.
// Pipeline: cvt -> Q/K proj -> V^T proj -> scores -> masked softmax -> AV -> out proj(+residual)
// GEMM staging uses gfx1250 async global->LDS (ASYNCcnt) with LDS double-buffer.
// ---------------------------------------------------------------------------

typedef __bf16 bf16_t;
typedef __attribute__((ext_vector_type(16))) __bf16 v16bf;
typedef __attribute__((ext_vector_type(8)))  float  v8f;
typedef int v4i __attribute__((vector_size(16)));   // GCC-style vector, matches builtin proto

#define B_ 4
#define S_ 2048
#define D_ 1024

#if __has_builtin(__builtin_amdgcn_global_load_async_to_lds_b128) && \
    __has_builtin(__builtin_amdgcn_s_wait_asynccnt)
#define USE_ASYNC_LDS 1
#else
#define USE_ASYNC_LDS 0
#endif

#define AS_GLOBAL(p) ((__attribute__((address_space(1))) v4i*)(p))
#define AS_LDS(p)    ((__attribute__((address_space(3))) v4i*)(p))

__device__ __forceinline__ bf16_t f2bf(float f) {
    unsigned u = __float_as_uint(f);
    unsigned r = u + 0x7FFFu + ((u >> 16) & 1u);   // round-to-nearest-even
    unsigned short h = (unsigned short)(r >> 16);
    return __builtin_bit_cast(bf16_t, h);
}

// ---------------------------------------------------------------------------
// fp32 -> bf16 conversion (grid-stride)
// ---------------------------------------------------------------------------
__global__ void __launch_bounds__(256) cvt_f32_to_bf16(const float* __restrict__ src,
                                                       bf16_t* __restrict__ dst, long long n) {
    long long i = (long long)blockIdx.x * blockDim.x + threadIdx.x;
    long long stride = (long long)gridDim.x * blockDim.x;
    for (; i < n; i += stride) dst[i] = f2bf(src[i]);
}

// ---------------------------------------------------------------------------
// C = A * Bt^T  (NT GEMM). A: MxK row-major bf16 (pitch lda). Bt: NxK row-major
// bf16 (pitch ldb) -> column n of B is the contiguous row n of Bt.
// Block tile 128x128, 8 waves (2x4), each wave 64x32 = 4x2 WMMA tiles, BK=32.
// LDS double-buffered; tiles staged with global_load_async_to_lds_b128.
// OUTF: 0 = bf16 out, 1 = f32 out, 2 = f32 out + residual
// biasMode: 0 = none, 1 = bias[n], 2 = bias[m]
// ---------------------------------------------------------------------------
template <int OUTF>
__global__ void __launch_bounds__(256, 1)
gemm_nt_wmma(const bf16_t* __restrict__ A,  long long lda, long long strideA,
             const bf16_t* __restrict__ Bt, long long ldb, long long strideB,
             void* __restrict__ Cv,         long long ldc, long long strideC,
             const float* __restrict__ bias, int biasMode,
             const float* __restrict__ resid,
             int M, int N, int K)
{
    constexpr int BM = 128, BN = 128, BK = 32, LDSP = 40;  // pitch 40 bf16 = 80B (16B aligned)
    __shared__ __align__(16) bf16_t As[2][BM * LDSP];
    __shared__ __align__(16) bf16_t Bs[2][BN * LDSP];

    const int tid  = threadIdx.x;
    const int lane = tid & 31;
    const int wave = tid >> 5;
    const int wr   = wave >> 2;   // 0..1 : M direction
    const int wc   = wave & 3;    // 0..3 : N direction
    const int m0   = blockIdx.y * BM;
    const int n0   = blockIdx.x * BN;
    const int b    = blockIdx.z;

    const bf16_t* Ab = A  + (size_t)b * strideA;
    const bf16_t* Bb = Bt + (size_t)b * strideB;

    // stage one 128x32 tile pair into LDS buffer `buf` for K-offset k0.
    // each thread: 4 chunks of A + 4 chunks of B (16B each) -> 8 async ops.
    auto stage = [&](int buf, int k0) {
        #pragma unroll
        for (int i = 0; i < 4; ++i) {
            int c    = i * 256 + tid;
            int row  = c >> 2;
            int col8 = (c & 3) * 8;
#if USE_ASYNC_LDS
            __builtin_amdgcn_global_load_async_to_lds_b128(
                AS_GLOBAL(Ab + (size_t)(m0 + row) * lda + k0 + col8),
                AS_LDS(&As[buf][row * LDSP + col8]), 0, 0);
            __builtin_amdgcn_global_load_async_to_lds_b128(
                AS_GLOBAL(Bb + (size_t)(n0 + row) * ldb + k0 + col8),
                AS_LDS(&Bs[buf][row * LDSP + col8]), 0, 0);
#else
            uint4 va = *(const uint4*)(Ab + (size_t)(m0 + row) * lda + k0 + col8);
            *(uint4*)&As[buf][row * LDSP + col8] = va;
            uint4 vb = *(const uint4*)(Bb + (size_t)(n0 + row) * ldb + k0 + col8);
            *(uint4*)&Bs[buf][row * LDSP + col8] = vb;
#endif
        }
    };

    v8f acc[4][2];
    #pragma unroll
    for (int i = 0; i < 4; ++i)
        #pragma unroll
        for (int j = 0; j < 2; ++j)
            acc[i][j] = v8f{};

    const int l15 = lane & 15;
    const int hi  = lane >> 4;        // 0 or 1
    const int ka  = hi * 8;           // A frag: K chunks [ka..ka+7] and [ka+16..ka+23]
    const int kb  = hi * 8;           // B frag: K chunks [kb..kb+7] and [kb+16..kb+23]

    const int nkt = K / BK;
    stage(0, 0);

    for (int kt = 0; kt < nkt; ++kt) {
        const int buf = kt & 1;
        if (kt + 1 < nkt) {
            stage(buf ^ 1, (kt + 1) * BK);     // overlap next tile's loads
#if USE_ASYNC_LDS
            __builtin_amdgcn_s_wait_asynccnt(8);  // 8 newest outstanding = next tile
#endif
        } else {
#if USE_ASYNC_LDS
            __builtin_amdgcn_s_wait_asynccnt(0);
#endif
        }
        __syncthreads();   // all waves' current-buffer loads complete

        union Frag { v16bf v; uint4 q[2]; };
        Frag af[4], bf[2];
        #pragma unroll
        for (int ms = 0; ms < 4; ++ms) {
            const bf16_t* p = &As[buf][(wr * 64 + ms * 16 + l15) * LDSP];
            af[ms].q[0] = *(const uint4*)(p + ka);
            af[ms].q[1] = *(const uint4*)(p + ka + 16);
        }
        #pragma unroll
        for (int nt = 0; nt < 2; ++nt) {
            const bf16_t* p = &Bs[buf][(wc * 32 + nt * 16 + l15) * LDSP];
            bf[nt].q[0] = *(const uint4*)(p + kb);
            bf[nt].q[1] = *(const uint4*)(p + kb + 16);
        }
        #pragma unroll
        for (int ms = 0; ms < 4; ++ms)
            #pragma unroll
            for (int nt = 0; nt < 2; ++nt)
                acc[ms][nt] = __builtin_amdgcn_wmma_f32_16x16x32_bf16(
                    false, af[ms].v, false, bf[nt].v, (short)0, acc[ms][nt], false, false);
        __syncthreads();   // protect the buffer overwritten by next iteration's stage
    }

    // epilogue: C layout -> VGPR r, lane L: M = r + (L>=16 ? 8 : 0), N = L&15
    const size_t cbase = (size_t)b * strideC;
    #pragma unroll
    for (int ms = 0; ms < 4; ++ms) {
        #pragma unroll
        for (int nt = 0; nt < 2; ++nt) {
            const int gn = n0 + wc * 32 + nt * 16 + l15;
            #pragma unroll
            for (int r = 0; r < 8; ++r) {
                const int gm = m0 + wr * 64 + ms * 16 + r + (hi << 3);
                float v = acc[ms][nt][r];
                if (biasMode == 1)      v += bias[gn];
                else if (biasMode == 2) v += bias[gm];
                const size_t idx = cbase + (size_t)gm * ldc + gn;
                if constexpr (OUTF == 0)      ((bf16_t*)Cv)[idx] = f2bf(v);
                else if constexpr (OUTF == 1) ((float*)Cv)[idx]  = v;
                else                          ((float*)Cv)[idx]  = v + resid[(size_t)gm * ldc + gn];
            }
        }
    }
}

// ---------------------------------------------------------------------------
// Row softmax with key-padding mask. One block per (b,q) row of 2048 fp32
// scores; writes bf16 probabilities in place (bf16 row pitch passed in).
// ---------------------------------------------------------------------------
__global__ void __launch_bounds__(256) softmax_mask_kernel(
        const float* __restrict__ scores,
        bf16_t* __restrict__ P, long long pitchP,
        const int* __restrict__ mask, float scale)
{
    const int q = blockIdx.x, b = blockIdx.y, tid = threadIdx.x;
    const float* row  = scores + ((size_t)b * S_ + q) * S_;
    bf16_t*      prow = P + ((size_t)b * S_ + q) * pitchP;
    const int*   mrow = mask + (size_t)b * S_;

    float vals[8];
    float vmax = -3.4e38f;
    #pragma unroll
    for (int i = 0; i < 8; ++i) {
        int j = tid + i * 256;
        float v = (mrow[j] == 0) ? -1e10f : row[j] * scale;
        vals[i] = v;
        vmax = fmaxf(vmax, v);
    }

    __shared__ float sred[256];
    sred[tid] = vmax; __syncthreads();
    #pragma unroll
    for (int s = 128; s > 0; s >>= 1) {
        if (tid < s) sred[tid] = fmaxf(sred[tid], sred[tid + s]);
        __syncthreads();
    }
    vmax = sred[0]; __syncthreads();

    float lsum = 0.f;
    #pragma unroll
    for (int i = 0; i < 8; ++i) { vals[i] = __expf(vals[i] - vmax); lsum += vals[i]; }
    sred[tid] = lsum; __syncthreads();
    #pragma unroll
    for (int s = 128; s > 0; s >>= 1) {
        if (tid < s) sred[tid] += sred[tid + s];
        __syncthreads();
    }
    const float inv = 1.0f / sred[0];

    // all fp32 reads of `row` happened before the reductions' barriers -> safe
    // to overwrite the row with bf16 now.
    #pragma unroll
    for (int i = 0; i < 8; ++i) prow[tid + i * 256] = f2bf(vals[i] * inv);
}

// ---------------------------------------------------------------------------
extern "C" void kernel_launch(void* const* d_in, const int* in_sizes, int n_in,
                              void* d_out, int out_size, void* d_ws, size_t ws_size,
                              hipStream_t stream)
{
    (void)in_sizes; (void)n_in; (void)out_size; (void)ws_size;

    const float* X    = (const float*)d_in[0];
    const int*   mask = (const int*)d_in[1];
    const float* Wq   = (const float*)d_in[2];
    const float* bq   = (const float*)d_in[3];
    const float* Wk   = (const float*)d_in[4];
    const float* bk   = (const float*)d_in[5];
    const float* Wv   = (const float*)d_in[6];
    const float* bv   = (const float*)d_in[7];
    const float* Wo   = (const float*)d_in[8];
    const float* bo   = (const float*)d_in[9];
    float* out = (float*)d_out;

    const size_t nX = (size_t)B_ * S_ * D_;   // 8M elements
    const size_t nW = (size_t)D_ * D_;        // 1M elements

    // workspace layout (bytes): total ~152 MB
    char* ws = (char*)d_ws;
    size_t off = 0;
    bf16_t* Xb  = (bf16_t*)(ws + off); off += nX * 2;               // 16 MB
    bf16_t* Wqb = (bf16_t*)(ws + off); off += nW * 2;               //  2 MB
    bf16_t* Wkb = (bf16_t*)(ws + off); off += nW * 2;
    bf16_t* Wvb = (bf16_t*)(ws + off); off += nW * 2;
    bf16_t* Wob = (bf16_t*)(ws + off); off += nW * 2;
    bf16_t* Qb  = (bf16_t*)(ws + off); off += nX * 2;               // 16 MB
    bf16_t* Kb  = (bf16_t*)(ws + off); off += nX * 2;               // 16 MB
    bf16_t* Vtb = (bf16_t*)(ws + off); off += nX * 2;               // 16 MB (V^T: B x D x S)
    float*  Sc  = (float*)(ws + off);  off += (size_t)B_ * S_ * S_ * 4; // 64 MB (P bf16 aliases)
    bf16_t* Ob  = (bf16_t*)(ws + off); off += nX * 2;               // 16 MB

    dim3 blk(256);

    // 1) fp32 -> bf16 conversions
    cvt_f32_to_bf16<<<dim3(2048), blk, 0, stream>>>(X,  Xb,  (long long)nX);
    cvt_f32_to_bf16<<<dim3(512),  blk, 0, stream>>>(Wq, Wqb, (long long)nW);
    cvt_f32_to_bf16<<<dim3(512),  blk, 0, stream>>>(Wk, Wkb, (long long)nW);
    cvt_f32_to_bf16<<<dim3(512),  blk, 0, stream>>>(Wv, Wvb, (long long)nW);
    cvt_f32_to_bf16<<<dim3(512),  blk, 0, stream>>>(Wo, Wob, (long long)nW);

    // 2) Q = X * Wq^T + bq ; K = X * Wk^T + bk    (M=8192, N=1024, K=1024)
    gemm_nt_wmma<0><<<dim3(D_/128, (B_*S_)/128, 1), blk, 0, stream>>>(
        Xb, D_, 0, Wqb, D_, 0, Qb, D_, 0, bq, 1, nullptr, B_*S_, D_, D_);
    gemm_nt_wmma<0><<<dim3(D_/128, (B_*S_)/128, 1), blk, 0, stream>>>(
        Xb, D_, 0, Wkb, D_, 0, Kb, D_, 0, bk, 1, nullptr, B_*S_, D_, D_);

    // 3) V^T[b] = Wv * X_b^T + bv (per-row bias)   (M=1024, N=2048, K=1024)
    gemm_nt_wmma<0><<<dim3(S_/128, D_/128, B_), blk, 0, stream>>>(
        Wvb, D_, 0, Xb, D_, (long long)S_*D_, Vtb, S_, (long long)D_*S_,
        bv, 2, nullptr, D_, S_, D_);

    // 4) Scores[b] = Q_b * K_b^T (raw fp32)        (M=N=2048, K=1024)
    gemm_nt_wmma<1><<<dim3(S_/128, S_/128, B_), blk, 0, stream>>>(
        Qb, D_, (long long)S_*D_, Kb, D_, (long long)S_*D_,
        Sc, S_, (long long)S_*S_, nullptr, 0, nullptr, S_, S_, D_);

    // 5) masked softmax rows -> bf16 P in place (bf16 row pitch = 2*S)
    softmax_mask_kernel<<<dim3(S_, B_), blk, 0, stream>>>(
        Sc, (bf16_t*)Sc, 2LL * S_, mask, 1.0f / 32.0f);

    // 6) O[b] = P_b * V_b                          (M=2048, N=1024, K=2048)
    gemm_nt_wmma<0><<<dim3(D_/128, S_/128, B_), blk, 0, stream>>>(
        (bf16_t*)Sc, 2LL*S_, 2LL*S_*S_, Vtb, S_, (long long)D_*S_,
        Ob, D_, (long long)S_*D_, nullptr, 0, nullptr, S_, D_, S_);

    // 7) out = O * Wo^T + bo + X (residual, fp32)  (M=8192, N=1024, K=1024)
    gemm_nt_wmma<2><<<dim3(D_/128, (B_*S_)/128, 1), blk, 0, stream>>>(
        Ob, D_, 0, Wob, D_, 0, out, D_, 0, bo, 1, X, B_*S_, D_, D_);
}
